// ModulatedDeformableConv2d_5669356831723
// MI455X (gfx1250) — matrix operand order repaired and checked by reading, not compiled, not run on hardware
//
#include <hip/hip_runtime.h>
#include <math.h>

typedef __attribute__((ext_vector_type(2)))  __bf16 bf16x2;
typedef __attribute__((ext_vector_type(8)))  __bf16 bf16x8;
typedef __attribute__((ext_vector_type(16))) __bf16 bf16x16;
typedef __attribute__((ext_vector_type(8)))  float  f32x8;

#define CH    64
#define IMH   128
#define IMW   128
#define TAPS  9
#define BROW  292   // u32 per B-LDS row: 576 bf16 data + 8 bf16 pad = 584 bf16 = 292 u32 (1168B = 73*16, bank-decorrelated)

static __device__ __forceinline__ unsigned pack_bf16x2(float a, float b) {
  // hardware f32->bf16 convert (v_cvt_pk_bf16_f32 class), RNE
  bf16x2 v;
  v[0] = (__bf16)a;
  v[1] = (__bf16)b;
  return __builtin_bit_cast(unsigned, v);
}

static __device__ __forceinline__ bf16x16 load_frag(const __bf16* p0, const __bf16* p1) {
  bf16x8 lo = *(const bf16x8*)p0;   // 16B-aligned ds_read_b128
  bf16x8 hi = *(const bf16x8*)p1;
  return __builtin_shufflevector(lo, hi, 0,1,2,3,4,5,6,7,8,9,10,11,12,13,14,15);
}

// ---------------- Kernel 1: 3x3 offset conv (N=27 padded to 32) + bias + sigmoid(mask) ----------------
__global__ __launch_bounds__(256) void offset_conv_kernel(
    const float* __restrict__ xe, const float* __restrict__ w_off,
    const float* __restrict__ b_off, float* __restrict__ om)
{
  __shared__ unsigned bsh[32 * BROW];        // B^T: [n][K], K = tap*64 + c, bf16 pairs
  __shared__ unsigned ash[8][16][36];        // per-wave A tile: 16 rows x 64 bf16 (+pad), 144B row stride

  const int tid  = threadIdx.x;
  const int wave = tid >> 5;
  const int lane = tid & 31;

  for (int i = tid; i < 32 * TAPS * 32; i += 256) {
    int cp = i & 31;
    int t  = (i >> 5) % TAPS;
    int n  = i / (TAPS * 32);
    float v0 = 0.f, v1 = 0.f;
    if (n < 27) {
      int c = cp * 2;
      v0 = w_off[(n * CH + c) * TAPS + t];
      v1 = w_off[(n * CH + c + 1) * TAPS + t];
    }
    bsh[n * BROW + t * 32 + cp] = pack_bf16x2(v0, v1);
  }
  __syncthreads();

  const int tile = blockIdx.x * 8 + wave;    // 8192 tiles of 16 pixels
  const int wo0  = (tile & 7) * 16;
  const int ho   = (tile >> 3) & (IMH - 1);
  const int b    = tile >> 10;

  const int row = lane & 15;
  const int g   = lane >> 4;

  f32x8 acc[2] = {};

  const int c0 = lane * 2;
  const float* xc0 = xe + (size_t)(b * CH + c0) * (IMH * IMW);

  for (int t = 0; t < TAPS; ++t) {
    const int ky = t / 3, kx = t % 3;
    const int y  = ho - 1 + ky;
    const bool vy = (y >= 0) && (y < IMH);
    const int yc = min(max(y, 0), IMH - 1);
    const float* prow = xc0 + (size_t)yc * IMW;
    #pragma unroll
    for (int r = 0; r < 16; ++r) {           // each lane fills channels {2*lane, 2*lane+1}, branchless
      const int xx  = wo0 - 1 + kx + r;
      const int xcl = min(max(xx, 0), IMW - 1);
      float v0 = prow[xcl];
      float v1 = prow[xcl + IMH * IMW];
      const bool ok = vy & (xx >= 0) & (xx < IMW);
      v0 = ok ? v0 : 0.f;
      v1 = ok ? v1 : 0.f;
      ash[wave][r][lane] = pack_bf16x2(v0, v1);
    }
    const __bf16* arow = (const __bf16*)&ash[wave][row][0];
    #pragma unroll
    for (int kc = 0; kc < 2; ++kc) {
      bf16x16 afrag = load_frag(arow + kc * 32 + 8 * g,
                                arow + kc * 32 + 16 + 8 * g);
      const int koff = t * 64 + kc * 32;
      #pragma unroll
      for (int nt = 0; nt < 2; ++nt) {
        const __bf16* brow = (const __bf16*)bsh + (nt * 16 + row) * (BROW * 2) + koff;
        bf16x16 bfrag = load_frag(brow + 8 * g, brow + 16 + 8 * g);
        acc[nt] = __builtin_amdgcn_wmma_f32_16x16x32_bf16(
            false, afrag, false, bfrag, (short)0, acc[nt], false, false);
      }
    }
  }

  #pragma unroll
  for (int nt = 0; nt < 2; ++nt) {
    const int n = nt * 16 + row;
    if (n < 27) {
      const float bv = b_off[n];
      float* dst = om + (((size_t)b * 27 + n) * IMH + ho) * IMW + wo0;
      #pragma unroll
      for (int j = 0; j < 8; ++j) {
        float v = acc[nt][j] + bv;
        if (n >= 18) v = 1.0f / (1.0f + __expf(-v));   // sigmoid only on mask channels
        dst[j + 8 * g] = v;
      }
    }
  }
}

// ---------------- Kernel 2: deformable bilinear gather + modulation + WMMA GEMM (N=64) ----------------
__global__ __launch_bounds__(256) void deform_conv_kernel(
    const float* __restrict__ x, const float* __restrict__ om,
    const float* __restrict__ weight, const float* __restrict__ bias,
    float* __restrict__ out)
{
  __shared__ unsigned bsh[64 * BROW];        // 73 KB
  __shared__ unsigned ash[8][16][36];        // 18 KB

  const int tid  = threadIdx.x;
  const int wave = tid >> 5;
  const int lane = tid & 31;

  for (int i = tid; i < 64 * TAPS * 32; i += 256) {
    int cp = i & 31;
    int t  = (i >> 5) % TAPS;
    int o  = i / (TAPS * 32);
    int c  = cp * 2;
    float v0 = weight[(o * CH + c) * TAPS + t];
    float v1 = weight[(o * CH + c + 1) * TAPS + t];
    bsh[o * BROW + t * 32 + cp] = pack_bf16x2(v0, v1);
  }
  __syncthreads();

  const int tile = blockIdx.x * 8 + wave;
  const int wo0  = (tile & 7) * 16;
  const int ho   = (tile >> 3) & (IMH - 1);
  const int b    = tile >> 10;

  const int row = lane & 15;
  const int g   = lane >> 4;

  f32x8 acc[4] = {};

  const int c0 = lane * 2;
  const float* xp0 = x + (size_t)(b * CH + c0) * (IMH * IMW);
  const float* xp1 = xp0 + IMH * IMW;
  const float* omb = om + (size_t)b * 27 * IMH * IMW;

  for (int t = 0; t < TAPS; ++t) {
    const int ky = t / 3, kx = t % 3;
    const float* offY = omb + ((size_t)t        * IMH + ho) * IMW + wo0;
    const float* offX = omb + ((size_t)(9 + t)  * IMH + ho) * IMW + wo0;
    const float* msk  = omb + ((size_t)(18 + t) * IMH + ho) * IMW + wo0;
    const float ybase = (float)(ho - 1 + ky);
    const float xbase = (float)(wo0 - 1 + kx);
    #pragma unroll 4
    for (int r = 0; r < 16; ++r) {
      const float py = offY[r] + ybase;                       // broadcast loads (same addr all lanes)
      const float px = offX[r] + xbase + (float)r;
      const float m  = msk[r];
      const float y0f = floorf(py), x0f = floorf(px);
      const int y0 = (int)y0f, x0 = (int)x0f;
      const float wy = py - y0f, wx = px - x0f;
      float w00 = (1.f - wy) * (1.f - wx);
      float w01 = (1.f - wy) * wx;
      float w10 = wy * (1.f - wx);
      float w11 = wy * wx;
      const bool vy0 = (y0 >= 0) & (y0 < IMH);
      const bool vy1 = (y0 + 1 >= 0) & (y0 + 1 < IMH);
      const bool vx0 = (x0 >= 0) & (x0 < IMW);
      const bool vx1 = (x0 + 1 >= 0) & (x0 + 1 < IMW);
      if (!(vy0 & vx0)) w00 = 0.f;
      if (!(vy0 & vx1)) w01 = 0.f;
      if (!(vy1 & vx0)) w10 = 0.f;
      if (!(vy1 & vx1)) w11 = 0.f;
      const int yc0 = min(max(y0, 0), IMH - 1);
      const int yc1 = min(max(y0 + 1, 0), IMH - 1);
      const int xc0 = min(max(x0, 0), IMW - 1);
      const int xc1 = min(max(x0 + 1, 0), IMW - 1);
      const int i00 = yc0 * IMW + xc0, i01 = yc0 * IMW + xc1;
      const int i10 = yc1 * IMW + xc0, i11 = yc1 * IMW + xc1;
      float s0 = w00 * xp0[i00] + w01 * xp0[i01] + w10 * xp0[i10] + w11 * xp0[i11];
      float s1 = w00 * xp1[i00] + w01 * xp1[i01] + w10 * xp1[i10] + w11 * xp1[i11];
      ash[wave][r][lane] = pack_bf16x2(s0 * m, s1 * m);
    }
    const __bf16* arow = (const __bf16*)&ash[wave][row][0];
    #pragma unroll
    for (int kc = 0; kc < 2; ++kc) {
      bf16x16 afrag = load_frag(arow + kc * 32 + 8 * g,
                                arow + kc * 32 + 16 + 8 * g);
      const int koff = t * 64 + kc * 32;
      #pragma unroll
      for (int nt = 0; nt < 4; ++nt) {
        const __bf16* brow = (const __bf16*)bsh + (nt * 16 + row) * (BROW * 2) + koff;
        bf16x16 bfrag = load_frag(brow + 8 * g, brow + 16 + 8 * g);
        acc[nt] = __builtin_amdgcn_wmma_f32_16x16x32_bf16(
            false, afrag, false, bfrag, (short)0, acc[nt], false, false);
      }
    }
  }

  #pragma unroll
  for (int nt = 0; nt < 4; ++nt) {
    const int n = nt * 16 + row;
    const float bv = bias[n];
    float* dst = out + (((size_t)b * CH + n) * IMH + ho) * IMW + wo0;
    #pragma unroll
    for (int j = 0; j < 8; ++j) {
      dst[j + 8 * g] = acc[nt][j] + bv;
    }
  }
}

extern "C" void kernel_launch(void* const* d_in, const int* in_sizes, int n_in,
                              void* d_out, int out_size, void* d_ws, size_t ws_size,
                              hipStream_t stream) {
  (void)in_sizes; (void)n_in; (void)out_size; (void)ws_size;
  const float* x       = (const float*)d_in[0];
  const float* x_extra = (const float*)d_in[1];
  const float* w_off   = (const float*)d_in[2];
  const float* b_off   = (const float*)d_in[3];
  const float* weight  = (const float*)d_in[4];
  const float* bias    = (const float*)d_in[5];
  float* om  = (float*)d_ws;           // 8*27*128*128 f32 = 14.2 MB scratch
  float* out = (float*)d_out;

  dim3 grid(1024), block(256);         // 8192 wave-tiles of 16 pixels, 8 waves/block
  offset_conv_kernel<<<grid, block, 0, stream>>>(x_extra, w_off, b_off, om);
  deform_conv_kernel<<<grid, block, 0, stream>>>(x, om, weight, bias, out);
}
